// LlamaAttention_57878979281332
// MI455X (gfx1250) — matrix operand order repaired and checked by reading, compiled, not run on hardware
//
#include <hip/hip_runtime.h>

// ---------------------------------------------------------------------------
// Llama attention layer for MI455X (gfx1250, wave32, WMMA + TDM).
// Pipeline: cvt(f32->f16) -> transpose weights -> WMMA GEMM (qkv, TDM-staged)
//           -> RoPE -> flash attention (WMMA) -> WMMA GEMM (o-proj, f32 out)
// ---------------------------------------------------------------------------

#define DEVI __device__ __forceinline__

typedef _Float16 half_t;
typedef __attribute__((ext_vector_type(16))) _Float16 v16h;
typedef __attribute__((ext_vector_type(8)))  _Float16 v8h;
typedef __attribute__((ext_vector_type(8)))  float    v8f;
typedef __attribute__((ext_vector_type(4)))  unsigned v4u;
typedef __attribute__((ext_vector_type(8)))  int      v8i;
typedef __attribute__((ext_vector_type(4)))  int      v4i;

#ifndef __has_builtin
#define __has_builtin(x) 0
#endif
#if defined(__AMDGCN__) && __has_builtin(__builtin_amdgcn_tensor_load_to_lds) && \
    __has_builtin(__builtin_amdgcn_s_wait_tensorcnt)
#define HAVE_TDM 1
#else
#define HAVE_TDM 0
#endif

constexpr int B_   = 4;
constexpr int S_   = 1024;
constexpr int HID_ = 4096;
constexpr int NH_  = 32;
constexpr int NKV_ = 8;
constexpr int D_   = 128;
constexpr int QKVN = (NH_ + 2 * NKV_) * D_;   // 6144
constexpr int MTOK = B_ * S_;                 // 4096
constexpr float QSCALE = 0.08838834764831845f; // 1/sqrt(128)

// A fragment (16x32 f16, M x K): lane m = lane&15, ksel = lane>>4.
// halves 0..15 = A[m][kbase+ksel*16+0..15].
DEVI v16h load_a_frag(const half_t* p) {
  v8h lo = *(const v8h*)(p);
  v8h hi = *(const v8h*)(p + 8);
  v16h r;
  #pragma unroll
  for (int i = 0; i < 8; ++i) { r[i] = lo[i]; r[i + 8] = hi[i]; }
  return r;
}

DEVI float rowmax16(float v) {
  #pragma unroll
  for (int m = 1; m < 16; m <<= 1) v = fmaxf(v, __shfl_xor(v, m, 32));
  return v;
}
DEVI float rowsum16(float v) {
  #pragma unroll
  for (int m = 1; m < 16; m <<= 1) v += __shfl_xor(v, m, 32);
  return v;
}

// ---------------------------------------------------------------------------
// Tensor Data Mover: 2D f16 tile (tile_w elems x tile_h rows) -> LDS.
// Builds the D# descriptor per CDNA5 ISA 08_async_tensor §8 and issues
// tensor_load_to_lds (TENSORcnt-tracked). One call = one wave-level DMA.
// ---------------------------------------------------------------------------
#if HAVE_TDM
DEVI void tdm_load_tile_2d(unsigned lds_off, const half_t* gptr,
                           unsigned tile_w, unsigned tile_h, unsigned stride_elems) {
  unsigned long long ga = (unsigned long long)(size_t)gptr;
  v4u g0;
  g0[0] = 1u;                                   // count=1 (valid user D#)
  g0[1] = lds_off;                              // lds_addr (bytes)
  g0[2] = (unsigned)ga;                         // global_addr[31:0]
  g0[3] = (unsigned)((ga >> 32) & 0x01FFFFFFu)  // global_addr[56:32]
          | (2u << 30);                         // type = 2 ("image")
  v8i g1;
  g1[0] = (int)(1u << 16);                      // data_size=1 (2 bytes), mask=0
  g1[1] = (int)((tile_w & 0xFFFFu) << 16);      // tensor_dim0[15:0] @ bits 63:48
  g1[2] = (int)((tile_w >> 16) | ((tile_h & 0xFFFFu) << 16));  // dim0 hi / dim1 lo
  g1[3] = (int)((tile_h >> 16) | ((tile_w & 0xFFFFu) << 16));  // dim1 hi / tile_dim0
  g1[4] = (int)(tile_h & 0xFFFFu);              // tile_dim1; tile_dim2=0 (2D)
  g1[5] = (int)stride_elems;                    // tensor_dim0_stride[31:0]
  g1[6] = 0;                                    // stride0 hi / stride1 lo
  g1[7] = 0;
  v4i z4 = {0, 0, 0, 0};
#if defined(__clang_major__) && __clang_major__ >= 23
  v8i z8 = {0, 0, 0, 0, 0, 0, 0, 0};
  __builtin_amdgcn_tensor_load_to_lds(g0, g1, z4, z4, z8, 0);
#else
  __builtin_amdgcn_tensor_load_to_lds(g0, g1, z4, z4, 0);
#endif
}
#endif

// ---------------------------------------------------------------------------
// Elementwise f32 -> f16
// ---------------------------------------------------------------------------
__global__ void __launch_bounds__(256)
cvt_f16_kernel(const float* __restrict__ x, half_t* __restrict__ y, int n) {
  int i = blockIdx.x * 256 + threadIdx.x;
  if (i < n) y[i] = (half_t)x[i];
}

// ---------------------------------------------------------------------------
// Tiled transpose + convert: WT[n][k] = (f16) W[k][n].   W: [K,N] f32.
// ---------------------------------------------------------------------------
__global__ void __launch_bounds__(256)
transpose_cvt(const float* __restrict__ W, half_t* __restrict__ WT, int K, int N) {
  __shared__ float tile[32][33];
  const int kt = blockIdx.x * 32, nt = blockIdx.y * 32;
  const int tx = threadIdx.x & 31, ty = threadIdx.x >> 5;
  #pragma unroll
  for (int r = ty; r < 32; r += 8)
    tile[r][tx] = W[(size_t)(kt + r) * N + nt + tx];
  __syncthreads();
  #pragma unroll
  for (int r = ty; r < 32; r += 8)
    WT[(size_t)(nt + r) * K + kt + tx] = (half_t)tile[tx][r];
}

// ---------------------------------------------------------------------------
// WMMA GEMM with TDM-staged, double-buffered LDS tiles.
// C[M,N] = A[M,K] * B[K,N], BT = B^T stored [N,K] f16.
// Block = 256 threads (8 waves), block tile 128x128, wave tile 64x32.
// Per k-step (K+=32): wave0 DMAs next A/B 128x32 tiles (8KB each) to LDS,
// waits TENSORcnt, all waves compute 8 WMMAs from the current buffers.
// ---------------------------------------------------------------------------
template <bool OUT_F16>
__global__ void __launch_bounds__(256)
gemm_wmma(const half_t* __restrict__ A, const half_t* __restrict__ BT,
          void* __restrict__ Cout, int M, int N, int K) {
  __shared__ alignas(32) half_t a_tile[2][128][32];
  __shared__ alignas(32) half_t b_tile[2][128][32];

  const int tid   = threadIdx.x;
  const int lane  = tid & 31;
  const int wave  = tid >> 5;
  const int mw    = wave >> 2;            // 0..1
  const int nw    = wave & 3;             // 0..3
  const int m0b   = blockIdx.y * 128;
  const int n0b   = blockIdx.x * 128;
  const int mlane = lane & 15;
  const int ksel  = lane >> 4;

  const half_t* Ab = A  + (size_t)m0b * K;
  const half_t* Bb = BT + (size_t)n0b * K;

  v8f acc[4][2];
  #pragma unroll
  for (int i = 0; i < 4; ++i)
    #pragma unroll
    for (int j = 0; j < 2; ++j) acc[i][j] = v8f{};

  // prologue: stage k-tile 0 into buffer 0
#if HAVE_TDM
  if (wave == 0) {
    tdm_load_tile_2d((unsigned)(size_t)&a_tile[0][0][0], Ab, 32, 128, (unsigned)K);
    tdm_load_tile_2d((unsigned)(size_t)&b_tile[0][0][0], Bb, 32, 128, (unsigned)K);
  }
#else
  { const int row = tid >> 1, hh = tid & 1;
    *(v16h*)&a_tile[0][row][hh * 16] = *(const v16h*)(Ab + (size_t)row * K + hh * 16);
    *(v16h*)&b_tile[0][row][hh * 16] = *(const v16h*)(Bb + (size_t)row * K + hh * 16);
  }
#endif

  int cur = 0;
  for (int kk = 0; kk < K; kk += 32, cur ^= 1) {
    // clamp next tile (redundant re-load of the last tile into the idle buffer)
    const int knext = (kk + 32 < K) ? (kk + 32) : kk;
#if HAVE_TDM
    if (wave == 0) {
      tdm_load_tile_2d((unsigned)(size_t)&a_tile[cur ^ 1][0][0], Ab + knext, 32, 128, (unsigned)K);
      tdm_load_tile_2d((unsigned)(size_t)&b_tile[cur ^ 1][0][0], Bb + knext, 32, 128, (unsigned)K);
      __builtin_amdgcn_s_wait_tensorcnt(2);   // the 2 DMAs for buf[cur] are done
    }
#else
    { const int row = tid >> 1, hh = tid & 1;
      *(v16h*)&a_tile[cur ^ 1][row][hh * 16] =
          *(const v16h*)(Ab + (size_t)row * K + knext + hh * 16);
      *(v16h*)&b_tile[cur ^ 1][row][hh * 16] =
          *(const v16h*)(Bb + (size_t)row * K + knext + hh * 16);
    }
#endif
    __syncthreads();   // buf[cur] ready for everyone

    v16h a[4], b[2];
    #pragma unroll
    for (int i = 0; i < 4; ++i)
      a[i] = load_a_frag(&a_tile[cur][mw * 64 + i * 16 + mlane][ksel * 16]);
    #pragma unroll
    for (int j = 0; j < 2; ++j)
      b[j] = *(const v16h*)&b_tile[cur][nw * 32 + j * 16 + mlane][ksel * 16];
    #pragma unroll
    for (int i = 0; i < 4; ++i)
      #pragma unroll
      for (int j = 0; j < 2; ++j)
        acc[i][j] = __builtin_amdgcn_wmma_f32_16x16x32_f16(
            false, a[i], false, b[j], (short)0, acc[i][j], false, false);

    __syncthreads();   // protect buf[cur^1] (next DMA/stores) vs this iter's reads
  }

  // C layout: element r of lane -> row = r + ksel*8, col = mlane (per tile)
  #pragma unroll
  for (int i = 0; i < 4; ++i)
    #pragma unroll
    for (int j = 0; j < 2; ++j)
      #pragma unroll
      for (int r = 0; r < 8; ++r) {
        const size_t row = (size_t)(m0b + mw * 64 + i * 16 + ksel * 8 + r);
        const size_t col = (size_t)(n0b + nw * 32 + j * 16 + mlane);
        if (OUT_F16) ((half_t*)Cout)[row * N + col] = (half_t)acc[i][j][r];
        else         ((float*)Cout)[row * N + col]  = acc[i][j][r];
      }
}

// ---------------------------------------------------------------------------
// RoPE (NeoX) + scatter into attention-friendly layouts.
//   q  [B,NH,S,D] (pre-scaled by 1/sqrt(D)),  k [B,NKV,S,D],  vT [B,NKV,D,S]
// ---------------------------------------------------------------------------
__global__ void __launch_bounds__(256)
rope_scatter(const half_t* __restrict__ qkv, const int* __restrict__ positions,
             half_t* __restrict__ q, half_t* __restrict__ k, half_t* __restrict__ vT) {
  const int t = blockIdx.x;
  const int b = t >> 10;
  const int s = t & (S_ - 1);
  const float pos = (float)positions[t];
  const half_t* row = qkv + (size_t)t * QKVN;

  for (int idx = threadIdx.x; idx < NH_ * 64; idx += 256) {
    const int h = idx >> 6, d = idx & 63;
    float x1 = (float)row[h * 128 + d];
    float x2 = (float)row[h * 128 + 64 + d];
    float fr = pos * __powf(10000.0f, -(float)d * (1.0f / 64.0f));
    float c, sn;
    __sincosf(fr, &sn, &c);
    half_t* qp = q + (((size_t)b * NH_ + h) * S_ + s) * D_;
    qp[d]      = (half_t)((x1 * c - x2 * sn) * QSCALE);
    qp[64 + d] = (half_t)((x2 * c + x1 * sn) * QSCALE);
  }
  for (int idx = threadIdx.x; idx < NKV_ * 64; idx += 256) {
    const int h = idx >> 6, d = idx & 63;
    float x1 = (float)row[NH_ * 128 + h * 128 + d];
    float x2 = (float)row[NH_ * 128 + h * 128 + 64 + d];
    float fr = pos * __powf(10000.0f, -(float)d * (1.0f / 64.0f));
    float c, sn;
    __sincosf(fr, &sn, &c);
    half_t* kp = k + (((size_t)b * NKV_ + h) * S_ + s) * D_;
    kp[d]      = (half_t)(x1 * c - x2 * sn);
    kp[64 + d] = (half_t)(x2 * c + x1 * sn);
  }
  for (int idx = threadIdx.x; idx < NKV_ * 128; idx += 256) {
    const int h = idx >> 7, d = idx & 127;
    vT[(((size_t)b * NKV_ + h) * D_ + d) * S_ + s] =
        row[(NH_ + NKV_) * 128 + h * 128 + d];
  }
}

// ---------------------------------------------------------------------------
// One kv block (32 columns) of flash attention. MASK=true only for the single
// diagonal block (q0 mod 32 in {0,16} => exactly one block needs masking).
// ---------------------------------------------------------------------------
template <bool MASK>
DEVI void attn_kv_block(int k0, int q0, int mlane, int ksel, int g8,
                        const half_t* __restrict__ kb, const half_t* __restrict__ vb,
                        const v16h (&qa)[4], half_t (*pshr)[32],
                        v8f (&o)[8], float (&mrow)[8], float (&lrow)[8]) {
  v8f s0 = v8f{}, s1 = v8f{};
  #pragma unroll
  for (int dc = 0; dc < 4; ++dc) {
    v16h b0 = *(const v16h*)(kb + (size_t)(k0      + mlane) * D_ + dc * 32 + ksel * 16);
    v16h b1 = *(const v16h*)(kb + (size_t)(k0 + 16 + mlane) * D_ + dc * 32 + ksel * 16);
    s0 = __builtin_amdgcn_wmma_f32_16x16x32_f16(false, qa[dc], false, b0, (short)0, s0, false, false);
    s1 = __builtin_amdgcn_wmma_f32_16x16x32_f16(false, qa[dc], false, b1, (short)0, s1, false, false);
  }
  #pragma unroll
  for (int r = 0; r < 8; ++r) {
    const int row = q0 + g8 + r;
    float v0 = s0[r], v1 = s1[r];
    if (MASK) {
      if (k0 + mlane      > row) v0 = -1e30f;
      if (k0 + 16 + mlane > row) v1 = -1e30f;
    }
    float mx    = rowmax16(fmaxf(v0, v1));
    float mnew  = fmaxf(mrow[r], mx);
    float p0    = __expf(v0 - mnew);
    float p1    = __expf(v1 - mnew);
    float alpha = __expf(mrow[r] - mnew);
    lrow[r] = lrow[r] * alpha + rowsum16(p0 + p1);
    mrow[r] = mnew;
    #pragma unroll
    for (int t = 0; t < 8; ++t) o[t][r] *= alpha;
    pshr[g8 + r][mlane]      = (half_t)p0;
    pshr[g8 + r][16 + mlane] = (half_t)p1;
  }
  __syncthreads();
  v16h pa = load_a_frag(&pshr[mlane][ksel * 16]);
  #pragma unroll
  for (int dt = 0; dt < 8; ++dt) {
    v16h bv = *(const v16h*)(vb + (size_t)(dt * 16 + mlane) * S_ + k0 + ksel * 16);
    o[dt] = __builtin_amdgcn_wmma_f32_16x16x32_f16(false, pa, false, bv, (short)0, o[dt], false, false);
  }
  __syncthreads();
}

// ---------------------------------------------------------------------------
// Flash attention: one wave per (b, h, 16-row q tile). Causal, GQA.
// Output: attn [B*S, NH*D] f16.
// ---------------------------------------------------------------------------
__global__ void __launch_bounds__(32)
attn_kernel(const half_t* __restrict__ q, const half_t* __restrict__ k,
            const half_t* __restrict__ vT, half_t* __restrict__ attn) {
  __shared__ alignas(32) half_t pshr[16][32];
  const int q0    = blockIdx.x * 16;
  const int h     = blockIdx.y;
  const int b     = blockIdx.z;
  const int kvh   = h >> 2;                // NH / NKV = 4
  const int lane  = threadIdx.x & 31;
  const int mlane = lane & 15;
  const int ksel  = lane >> 4;
  const int g8    = ksel * 8;

  const half_t* qb = q  + (((size_t)b * NH_  + h  ) * S_ + q0) * D_;
  const half_t* kb = k  + (((size_t)b * NKV_ + kvh) * S_      ) * D_;
  const half_t* vb = vT + (((size_t)b * NKV_ + kvh) * D_      ) * S_;

  v16h qa[4];
  #pragma unroll
  for (int dc = 0; dc < 4; ++dc)
    qa[dc] = load_a_frag(qb + (size_t)mlane * D_ + dc * 32 + ksel * 16);

  v8f o[8];
  #pragma unroll
  for (int t = 0; t < 8; ++t) o[t] = v8f{};
  float mrow[8], lrow[8];
  #pragma unroll
  for (int r = 0; r < 8; ++r) { mrow[r] = -1e30f; lrow[r] = 0.0f; }

  const int kmain = (q0 / 32) * 32;   // blocks [0,kmain) are fully unmasked
  for (int k0 = 0; k0 < kmain; k0 += 32)
    attn_kv_block<false>(k0, q0, mlane, ksel, g8, kb, vb, qa, pshr, o, mrow, lrow);
  attn_kv_block<true>(kmain, q0, mlane, ksel, g8, kb, vb, qa, pshr, o, mrow, lrow);

  half_t* ob = attn + (size_t)(b * S_ + q0) * (NH_ * D_) + h * D_;
  #pragma unroll
  for (int r = 0; r < 8; ++r) {
    const float inv = 1.0f / lrow[r];
    #pragma unroll
    for (int t = 0; t < 8; ++t)
      ob[(size_t)(g8 + r) * (NH_ * D_) + t * 16 + mlane] = (half_t)(o[t][r] * inv);
  }
}

// ---------------------------------------------------------------------------
// Host launcher
// ---------------------------------------------------------------------------
extern "C" void kernel_launch(void* const* d_in, const int* in_sizes, int n_in,
                              void* d_out, int out_size, void* d_ws, size_t ws_size,
                              hipStream_t stream) {
  (void)in_sizes; (void)n_in; (void)out_size; (void)ws_size;
  const int*   positions = (const int*)d_in[0];
  const float* hidden    = (const float*)d_in[1];
  const float* w_qkv     = (const float*)d_in[2];
  const float* w_o       = (const float*)d_in[3];
  float*       out       = (float*)d_out;

  char* ws = (char*)d_ws;
  const size_t MB = 1ull << 20;
  // Region map (overlaid across pipeline stages, 160 MB total):
  //  [  0, 48): wqkvT (gemm1)        -> then q(32MB)+k(8MB)+vT(8MB)
  //  [ 48, 80): woT                  (live until gemm2)
  //  [ 80,112): Xh (gemm1 input)     -> then attn output (32MB)
  //  [112,160): qkv (gemm1 output, rope input)
  half_t* wqkvT = (half_t*)(ws + 0);
  half_t* qbuf  = (half_t*)(ws + 0);
  half_t* kbuf  = (half_t*)(ws + 32 * MB);
  half_t* vTbuf = (half_t*)(ws + 40 * MB);
  half_t* woT   = (half_t*)(ws + 48 * MB);
  half_t* Xh    = (half_t*)(ws + 80 * MB);
  half_t* attn  = (half_t*)(ws + 80 * MB);
  half_t* qkv   = (half_t*)(ws + 112 * MB);

  cvt_f16_kernel<<<(MTOK * HID_) / 256, 256, 0, stream>>>(hidden, Xh, MTOK * HID_);
  transpose_cvt<<<dim3(HID_ / 32, QKVN / 32), 256, 0, stream>>>(w_qkv, wqkvT, HID_, QKVN);
  transpose_cvt<<<dim3(HID_ / 32, HID_ / 32), 256, 0, stream>>>(w_o, woT, HID_, HID_);
  gemm_wmma<true><<<dim3(QKVN / 128, MTOK / 128), 256, 0, stream>>>(
      Xh, wqkvT, (void*)qkv, MTOK, QKVN, HID_);
  rope_scatter<<<MTOK, 256, 0, stream>>>(qkv, positions, qbuf, kbuf, vTbuf);
  attn_kernel<<<dim3(S_ / 16, NH_, B_), 32, 0, stream>>>(qbuf, kbuf, vTbuf, attn);
  gemm_wmma<false><<<dim3(HID_ / 128, MTOK / 128), 256, 0, stream>>>(
      attn, woT, (void*)out, MTOK, HID_, HID_);
}
